// VectorQuantizer_61632780697603
// MI455X (gfx1250) — matrix operand order repaired and checked by reading, not compiled, not run on hardware
//
#include <hip/hip_runtime.h>
#include <hip/hip_fp16.h>

typedef __attribute__((ext_vector_type(16))) _Float16 v16h;
typedef __attribute__((ext_vector_type(8)))  float    v8f;

#define N_TOT   65536          // 64*32*32 rows of z_flat
#define CDIM    256            // latent dim
#define KCB     512            // codebook size
#define HWDIM   1024           // 32*32
#define LDS_ROW 264            // padded halves per codebook row (528B = 132 dwords -> bank-conflict free)
#define QSZ     ((size_t)N_TOT * CDIM)   // quantized output elements

union AFrag { v16h v; uint4 q[2]; };

// ---------------------------------------------------------------------------
// Kernel 1: codebook f32 -> f16 (row major), squared norms, zero the loss slot
// ---------------------------------------------------------------------------
__global__ void prep_codebook(const float* __restrict__ cb,
                              _Float16* __restrict__ cb16,
                              float* __restrict__ e2,
                              float* __restrict__ loss_slot) {
    __shared__ float red[256];
    int row = blockIdx.x;
    int t   = threadIdx.x;
    float v = cb[row * CDIM + t];
    cb16[row * CDIM + t] = (_Float16)v;
    red[t] = v * v;
    __syncthreads();
    for (int s = 128; s > 0; s >>= 1) {
        if (t < s) red[t] += red[t + s];
        __syncthreads();
    }
    if (t == 0) e2[row] = red[0];
    if (row == 0 && t == 0) *loss_slot = 0.0f;
}

// ---------------------------------------------------------------------------
// Kernel 2: WMMA distance + argmin, A loaded straight from NCHW z.
//   Block = 256 threads (8 waves). Wave handles 32 rows (two 16x16 M-tiles).
//   A-fragment layout (ISA 7.12.2): lane = M = hw (contiguous in z!), so the
//   16 lanes of a half-group read 16 consecutive floats of z[b,c,:] -> each
//   global_load_b32 consumes two full 64B segments. f32->f16 RNE on the fly.
//   Full f16 codebook staged in LDS (264-half padded rows) + norms.
//   dist = |e|^2 - 2*(z.e)   (|z|^2 constant per row -> dropped)
// ---------------------------------------------------------------------------
__global__ void __launch_bounds__(256)
vq_argmin(const float* __restrict__ z,
          const _Float16* __restrict__ cb16,
          const float* __restrict__ e2g,
          int* __restrict__ idx_out) {
    extern __shared__ __align__(16) char smem[];
    _Float16* scb = (_Float16*)smem;                               // [512][264]
    float*    se2 = (float*)(smem + KCB * LDS_ROW * sizeof(_Float16));

    int tid = threadIdx.x;

    // Stage codebook (256 KB) global -> LDS in 16B chunks, row padding applied per-chunk
    const uint4* src = (const uint4*)cb16;          // 32 chunks of 8 halves per row
    for (int chunk = tid; chunk < KCB * 32; chunk += 256) {
        int row = chunk >> 5;
        int c8  = chunk & 31;
        *(uint4*)(scb + row * LDS_ROW + c8 * 8) = src[chunk];
    }
    for (int i = tid; i < KCB; i += 256) se2[i] = e2g[i];
    __syncthreads();

    int lane = tid & 31;
    int half = lane >> 4;          // lane group 0/1
    int l16  = lane & 15;
    int wave = tid >> 5;

    int nblk = blockIdx.x * 256;   // 256 rows per block, all within one batch image
    int bb   = nblk >> 10;
    int hw_w = (nblk & 1023) + wave * 32;
    int n0   = nblk + wave * 32;

    // A fragments: 16x32 f16 per chunk; lane = M; K = 8*half + j (lo) / 16+8*half+j (hi)
    AFrag a[2][8];
    #pragma unroll
    for (int mt = 0; mt < 2; ++mt) {
        const float* zp = z + ((size_t)bb * CDIM) * HWDIM + (hw_w + mt * 16 + l16);
        #pragma unroll
        for (int cc = 0; cc < 8; ++cc) {
            #pragma unroll
            for (int j = 0; j < 8; ++j) {
                int c0 = cc * 32 + half * 8 + j;        // K = 8h..8h+7
                int c1 = cc * 32 + 16 + half * 8 + j;   // K = 16+8h..16+8h+7
                a[mt][cc].v[j]     = (_Float16)zp[(size_t)c0 * HWDIM];
                a[mt][cc].v[j + 8] = (_Float16)zp[(size_t)c1 * HWDIM];
            }
        }
    }

    float mind0[8], mind1[8];
    int   mini0[8], mini1[8];
    #pragma unroll
    for (int r = 0; r < 8; ++r) {
        mind0[r] = 3.4e38f; mind1[r] = 3.4e38f; mini0[r] = 0; mini1[r] = 0;
    }

    for (int kt = 0; kt < 32; ++kt) {
        int col = kt * 16 + l16;                     // codebook entry owned by this lane
        const _Float16* brow = scb + col * LDS_ROW + half * 8;   // B: lane=N, K=8*half+j (lo) / 16+8*half+j (hi)
        v8f acc0 = {};
        v8f acc1 = {};
        #pragma unroll
        for (int cc = 0; cc < 8; ++cc) {
            AFrag b;
            b.q[0] = *(const uint4*)(brow + cc * 32);
            b.q[1] = *(const uint4*)(brow + cc * 32 + 16);
            acc0 = __builtin_amdgcn_wmma_f32_16x16x32_f16(false, a[0][cc].v, false, b.v,
                                                          (short)0, acc0, false, false);
            acc1 = __builtin_amdgcn_wmma_f32_16x16x32_f16(false, a[1][cc].v, false, b.v,
                                                          (short)0, acc1, false, false);
        }
        float e2v = se2[col];
        #pragma unroll
        for (int r = 0; r < 8; ++r) {
            float d0 = fmaf(-2.0f, acc0[r], e2v);
            float d1 = fmaf(-2.0f, acc1[r], e2v);
            if (d0 < mind0[r]) { mind0[r] = d0; mini0[r] = col; }
            if (d1 < mind1[r]) { mind1[r] = d1; mini1[r] = col; }
        }
    }

    // Reduce (val,idx) across the 16 lanes holding row M = r + 8*half; first-index ties.
    #pragma unroll
    for (int mt = 0; mt < 2; ++mt) {
        #pragma unroll
        for (int r = 0; r < 8; ++r) {
            float v = mt ? mind1[r] : mind0[r];
            int   i = mt ? mini1[r] : mini0[r];
            #pragma unroll
            for (int off = 8; off > 0; off >>= 1) {
                float vo = __shfl_xor(v, off, 16);
                int   io = __shfl_xor(i, off, 16);
                if (vo < v || (vo == v && io < i)) { v = vo; i = io; }
            }
            if (l16 == 0) idx_out[n0 + mt * 16 + r + 8 * half] = i;
        }
    }
}

// ---------------------------------------------------------------------------
// Kernel 3: gather z_q, write quantized = z + (z_q - z) (matches JAX expr),
//           indices (as float), and 1.25*mean((z_q - z)^2) loss.
//   Grid: 4 c-chunks x 256 n-tiles; coalesced on hw for z/out.
// ---------------------------------------------------------------------------
__global__ void vq_output(const float* __restrict__ z,
                          const float* __restrict__ cb,
                          const int* __restrict__ idx_ws,
                          float* __restrict__ out) {
    __shared__ float red[256];
    int t   = threadIdx.x;
    int hb  = blockIdx.x & 255;   // n tile
    int cbk = blockIdx.x >> 8;    // 0..3 c chunk
    int n   = hb * 256 + t;
    int bb  = n >> 10;
    int hw  = n & 1023;
    int idx = idx_ws[n];
    const float* crow = cb + (size_t)idx * CDIM;

    float lsum = 0.0f;
    for (int c = cbk * 64; c < cbk * 64 + 64; ++c) {
        size_t zoff = ((size_t)bb * CDIM + c) * HWDIM + hw;
        float zq = crow[c];
        float zv = z[zoff];
        float d  = zq - zv;
        out[zoff] = zv + d;       // straight-through value, same rounding as reference
        lsum += d * d;
    }
    if (cbk == 0) out[QSZ + n] = (float)idx;

    red[t] = lsum;
    __syncthreads();
    for (int s = 128; s > 0; s >>= 1) {
        if (t < s) red[t] += red[t + s];
        __syncthreads();
    }
    if (t == 0)
        atomicAdd(&out[QSZ + N_TOT], red[0] * (1.25f / (float)QSZ));
}

// ---------------------------------------------------------------------------
extern "C" void kernel_launch(void* const* d_in, const int* in_sizes, int n_in,
                              void* d_out, int out_size, void* d_ws, size_t ws_size,
                              hipStream_t stream) {
    (void)in_sizes; (void)n_in; (void)out_size; (void)ws_size;
    const float* z  = (const float*)d_in[0];
    const float* cb = (const float*)d_in[1];

    char* ws = (char*)d_ws;
    _Float16* cb16 = (_Float16*)ws;                                  // 256 KB
    float*    e2   = (float*)(ws + (size_t)KCB * CDIM * 2);          // 2 KB
    int*      idxw = (int*)((char*)e2 + KCB * sizeof(float));        // 256 KB

    float* outf  = (float*)d_out;
    float* lossp = outf + QSZ + N_TOT;

    prep_codebook<<<KCB, 256, 0, stream>>>(cb, cb16, e2, lossp);

    size_t lds_bytes = (size_t)KCB * LDS_ROW * sizeof(_Float16) + KCB * sizeof(float); // 272384
    vq_argmin<<<N_TOT / 256, 256, lds_bytes, stream>>>(z, cb16, e2, idxw);

    vq_output<<<4 * 256, 256, 0, stream>>>(z, cb, idxw, outf);
}
